// MultiHeadCrossAttention_5188320493947
// MI455X (gfx1250) — compile-verified, matrix operand order checked by reading, not run on hardware
//
#include <hip/hip_runtime.h>
#include <hip/hip_bf16.h>
#include <cstdint>

// ---------------------------------------------------------------------------
// MultiHeadCrossAttention for MI455X (gfx1250, wave32, WMMA 16x16x32 f16)
// B=2, Lq=1024, Lk=2048, D=1024, H=16, dh=64
// d_out = [ out (2*1024*1024 f32) | attn (2*16*1024*2048 f32) ]
// ---------------------------------------------------------------------------

#define B_SZ   2
#define LQ     1024
#define LK     2048
#define DMODEL 1024
#define NHEADS 16
#define DH     64

typedef __attribute__((ext_vector_type(16))) _Float16 v16h;
typedef __attribute__((ext_vector_type(2)))  __fp16   v2fp16;   // cvt_pkrtz return type
typedef __attribute__((ext_vector_type(8)))  float    v8f;

#define KB   32            // K-depth per WMMA step
#define LDH  48            // padded LDS row stride in halves (96 B, 16-B aligned)

// ---- WMMA wrapper ----------------------------------------------------------
__device__ __forceinline__ v8f wmma_f16(v16h a, v16h b, v8f c) {
    return __builtin_amdgcn_wmma_f32_16x16x32_f16(
        false, a, false, b, (short)0, c, false, false);
}

// ---- Fragment loader: 16 rows x 32 K from LDS (ISA 16-bit A layout) --------
__device__ __forceinline__ v16h lds_load_frag(const _Float16* base) {
    const int lane = threadIdx.x & 31;
    const int r    = lane & 15;
    const int koff = (lane & 16) ? 8 : 0;
    const _Float16* p = base + r * LDH + koff;
    v16h f;
#pragma unroll
    for (int g = 0; g < 8; ++g) {
        const int k = ((g & 4) ? 16 : 0) + 2 * (g & 3);
        f[2 * g]     = p[k];
        f[2 * g + 1] = p[k + 1];
    }
    return f;
}

// ---- Staging: 64x32 f32 tile -> f16 LDS tile (convert path, 128 thr) -------
__device__ __forceinline__ void stage_f32(const float* __restrict__ src, int ld,
                                          int row0, int k0,
                                          _Float16 (*dst)[LDH], int tid) {
#pragma unroll
    for (int it = 0; it < 4; ++it) {
        const int idx = tid + it * 128;          // 512 chunks of 4 floats
        const int r   = idx >> 3;                // 8 chunks per 32-col row
        const int c   = (idx & 7) * 4;
        const float4 v = *(const float4*)(src + (size_t)(row0 + r) * ld + k0 + c);
        union { v2fp16 h2[2]; unsigned long long u; } u;
        u.h2[0] = __builtin_amdgcn_cvt_pkrtz(v.x, v.y);
        u.h2[1] = __builtin_amdgcn_cvt_pkrtz(v.z, v.w);
        *(unsigned long long*)&dst[r][c] = u.u;
    }
}

// ---- Staging: 64x32 f16 tile -> LDS via async copy engine (ASYNCcnt) -------
// Each lane fires two fire-and-forget 16-byte global->LDS copies; no VGPR
// round-trip. LDS byte address = low 32 bits of the flat shared pointer
// (LDS aperture truncation rule).
__device__ __forceinline__ void stage_f16_async(const _Float16* __restrict__ src, int ld,
                                                int row0, int k0,
                                                _Float16 (*dst)[LDH], int tid) {
#if defined(__gfx1250__)
#pragma unroll
    for (int it = 0; it < 2; ++it) {
        const int idx = tid + it * 128;          // 256 chunks of 8 halves
        const int r   = idx >> 2;                // 4 chunks per 32-col row
        const int c   = (idx & 3) * 8;
        const unsigned long long g =
            (unsigned long long)(uintptr_t)(src + (size_t)(row0 + r) * ld + k0 + c);
        const unsigned int l = (unsigned int)(uintptr_t)&dst[r][c];
        asm volatile("global_load_async_to_lds_b128 %0, %1, off"
                     :: "v"(l), "v"(g) : "memory");
    }
#else
#pragma unroll
    for (int it = 0; it < 2; ++it) {
        const int idx = tid + it * 128;
        const int r   = idx >> 2;
        const int c   = (idx & 3) * 8;
        const uint4 v = *(const uint4*)(src + (size_t)(row0 + r) * ld + k0 + c);
        *(uint4*)&dst[r][c] = v;
    }
#endif
}

__device__ __forceinline__ void wait_async_lds() {
#if defined(__gfx1250__)
    asm volatile("s_wait_asynccnt 0x0" ::: "memory");
#endif
}

// ---------------------------------------------------------------------------
// Projection GEMM:  C[M,N] = A[M,K](f32) * B[N,K]^T(f32) + bias[N]
// 128 thr = 4 waves; macro tile 64x64; wave = 32x32; double-buffered LDS.
// MODE 0: f32 row-major   MODE 1: f16 [B,H,L,dh]   MODE 3: f16 [B,H,dh,L]
// ---------------------------------------------------------------------------
template <int MODE, int LSHIFT>
__global__ __launch_bounds__(128)
void gemm_proj(const float* __restrict__ A, const float* __restrict__ Bm,
               const float* __restrict__ bias,
               float* __restrict__ outF32, _Float16* __restrict__ outF16,
               int M, int N, int K) {
    __shared__ _Float16 sA[2][64][LDH];
    __shared__ _Float16 sB[2][64][LDH];

    const int tid   = threadIdx.x;
    const int wave  = tid >> 5;
    const int lane  = tid & 31;
    const int wr    = (wave >> 1) * 32;
    const int wc    = (wave & 1) * 32;
    const int tileM = blockIdx.x * 64;
    const int tileN = blockIdx.y * 64;

    v8f acc00 = {}, acc01 = {}, acc10 = {}, acc11 = {};

    const int nIter = K / KB;
    stage_f32(A,  K, tileM, 0, sA[0], tid);
    stage_f32(Bm, K, tileN, 0, sB[0], tid);
    int cur = 0;
    for (int it = 0; it < nIter; ++it) {
        __syncthreads();
        if (it + 1 < nIter) {
            stage_f32(A,  K, tileM, (it + 1) * KB, sA[cur ^ 1], tid);
            stage_f32(Bm, K, tileN, (it + 1) * KB, sB[cur ^ 1], tid);
        }
        const v16h a0 = lds_load_frag(&sA[cur][wr][0]);
        const v16h a1 = lds_load_frag(&sA[cur][wr + 16][0]);
        const v16h b0 = lds_load_frag(&sB[cur][wc][0]);
        const v16h b1 = lds_load_frag(&sB[cur][wc + 16][0]);
        acc00 = wmma_f16(a0, b0, acc00);
        acc01 = wmma_f16(a0, b1, acc01);
        acc10 = wmma_f16(a1, b0, acc10);
        acc11 = wmma_f16(a1, b1, acc11);
        cur ^= 1;
    }

    const int colL   = lane & 15;
    const int rowAdd = (lane & 16) ? 8 : 0;
#pragma unroll
    for (int i = 0; i < 2; ++i) {
#pragma unroll
        for (int j = 0; j < 2; ++j) {
            const v8f acc = (i == 0) ? (j == 0 ? acc00 : acc01)
                                     : (j == 0 ? acc10 : acc11);
#pragma unroll
            for (int r = 0; r < 8; ++r) {
                const int m = tileM + wr + i * 16 + rowAdd + r;
                const int n = tileN + wc + j * 16 + colL;
                const float v = acc[r] + bias[n];
                if (MODE == 0) {
                    outF32[(size_t)m * DMODEL + n] = v;
                } else {
                    const int b = m >> LSHIFT;
                    const int l = m & ((1 << LSHIFT) - 1);
                    const int h = n >> 6, d = n & 63;          // dh = 64
                    size_t off;
                    if (MODE == 3)
                        off = (((size_t)(b * NHEADS + h) * DH + d) << LSHIFT) + l;
                    else
                        off = ((((size_t)(b * NHEADS + h) << LSHIFT) + l) * DH) + d;
                    outF16[off] = (_Float16)v;
                }
            }
        }
    }
}

// ---------------------------------------------------------------------------
// Scores:  S = Qh.Kh^T / 8 + bias ; mask==0 -> -1e9 ; f32 into attn region.
// grid = (Lq/64, Lk/64, B*H), block = 128.  K = dh = 64 (2 steps).
// Qh/Kh tiles staged with async global->LDS copies.
// ---------------------------------------------------------------------------
__global__ __launch_bounds__(128)
void scores_kernel(const _Float16* __restrict__ Qh, const _Float16* __restrict__ Kh,
                   const float* __restrict__ bias, const int* __restrict__ mask,
                   float* __restrict__ P) {
    __shared__ _Float16 sA[2][64][LDH];
    __shared__ _Float16 sB[2][64][LDH];

    const int bh = blockIdx.z;
    const int b  = bh >> 4;
    const _Float16* Qp = Qh + (size_t)bh * LQ * DH;
    const _Float16* Kp = Kh + (size_t)bh * LK * DH;

    const int tid   = threadIdx.x;
    const int wave  = tid >> 5;
    const int lane  = tid & 31;
    const int wr    = (wave >> 1) * 32;
    const int wc    = (wave & 1) * 32;
    const int tileM = blockIdx.x * 64;   // q
    const int tileN = blockIdx.y * 64;   // k

    v8f acc00 = {}, acc01 = {}, acc10 = {}, acc11 = {};

    stage_f16_async(Qp, DH, tileM, 0, sA[0], tid);
    stage_f16_async(Kp, DH, tileN, 0, sB[0], tid);
    int cur = 0;
#pragma unroll
    for (int it = 0; it < DH / KB; ++it) {
        wait_async_lds();
        __syncthreads();
        if (it + 1 < DH / KB) {
            stage_f16_async(Qp, DH, tileM, (it + 1) * KB, sA[cur ^ 1], tid);
            stage_f16_async(Kp, DH, tileN, (it + 1) * KB, sB[cur ^ 1], tid);
        }
        const v16h a0 = lds_load_frag(&sA[cur][wr][0]);
        const v16h a1 = lds_load_frag(&sA[cur][wr + 16][0]);
        const v16h b0 = lds_load_frag(&sB[cur][wc][0]);
        const v16h b1 = lds_load_frag(&sB[cur][wc + 16][0]);
        acc00 = wmma_f16(a0, b0, acc00);
        acc01 = wmma_f16(a0, b1, acc01);
        acc10 = wmma_f16(a1, b0, acc10);
        acc11 = wmma_f16(a1, b1, acc11);
        cur ^= 1;
    }

    const float scale = 0.125f;          // 1/sqrt(64)
    const int colL   = lane & 15;
    const int rowAdd = (lane & 16) ? 8 : 0;
#pragma unroll
    for (int i = 0; i < 2; ++i) {
#pragma unroll
        for (int j = 0; j < 2; ++j) {
            const v8f acc = (i == 0) ? (j == 0 ? acc00 : acc01)
                                     : (j == 0 ? acc10 : acc11);
#pragma unroll
            for (int r = 0; r < 8; ++r) {
                const int q = tileM + wr + i * 16 + rowAdd + r;
                const int k = tileN + wc + j * 16 + colL;
                const size_t off = ((size_t)bh * LQ + q) * LK + k;
                float v = acc[r] * scale + bias[off];
                if (mask[((size_t)b * LQ + q) * LK + k] == 0) v = -1e9f;
                P[off] = v;
            }
        }
    }
}

// ---------------------------------------------------------------------------
// Row softmax in place. One 256-thread block per row of 2048.
// ---------------------------------------------------------------------------
__global__ __launch_bounds__(256)
void softmax_rows(float* __restrict__ P) {
    float* p = P + (size_t)blockIdx.x * LK;
    __shared__ float red[256];
    const int tid = threadIdx.x;

    float m = -3.4e38f;
    for (int c = tid; c < LK; c += 256) m = fmaxf(m, p[c]);
    red[tid] = m;
    __syncthreads();
    for (int s = 128; s > 0; s >>= 1) {
        if (tid < s) red[tid] = fmaxf(red[tid], red[tid + s]);
        __syncthreads();
    }
    m = red[0];
    __syncthreads();

    float sum = 0.0f;
    for (int c = tid; c < LK; c += 256) {
        const float e = __expf(p[c] - m);
        p[c] = e;
        sum += e;
    }
    red[tid] = sum;
    __syncthreads();
    for (int s = 128; s > 0; s >>= 1) {
        if (tid < s) red[tid] += red[tid + s];
        __syncthreads();
    }
    const float inv = 1.0f / red[0];
    for (int c = tid; c < LK; c += 256) p[c] *= inv;
}

// ---------------------------------------------------------------------------
// O[b,h,q,d] = sum_k P[b,h,q,k] * Vt[b,h,d,k] -> heads-merged f32 [B,Lq,D]
// grid = (Lq/64, 1, B*H), block = 128.  Vt staged via async copies.
// ---------------------------------------------------------------------------
__global__ __launch_bounds__(128)
void av_kernel(const float* __restrict__ P, const _Float16* __restrict__ Vt,
               float* __restrict__ Oheads) {
    __shared__ _Float16 sA[2][64][LDH];
    __shared__ _Float16 sB[2][64][LDH];

    const int bh = blockIdx.z;
    const int b  = bh >> 4;
    const int h  = bh & 15;
    const float*    Pp = P  + (size_t)bh * LQ * LK;
    const _Float16* Vp = Vt + (size_t)bh * DH * LK;

    const int tid   = threadIdx.x;
    const int wave  = tid >> 5;
    const int lane  = tid & 31;
    const int wr    = (wave >> 1) * 32;
    const int wc    = (wave & 1) * 32;
    const int tileM = blockIdx.x * 64;   // q

    v8f acc00 = {}, acc01 = {}, acc10 = {}, acc11 = {};

    stage_f32(Pp, LK, tileM, 0, sA[0], tid);
    stage_f16_async(Vp, LK, 0, 0, sB[0], tid);
    int cur = 0;
    for (int it = 0; it < LK / KB; ++it) {
        wait_async_lds();
        __syncthreads();
        if (it + 1 < LK / KB) {
            stage_f32(Pp, LK, tileM, (it + 1) * KB, sA[cur ^ 1], tid);
            stage_f16_async(Vp, LK, 0, (it + 1) * KB, sB[cur ^ 1], tid);
        }
        const v16h a0 = lds_load_frag(&sA[cur][wr][0]);
        const v16h a1 = lds_load_frag(&sA[cur][wr + 16][0]);
        const v16h b0 = lds_load_frag(&sB[cur][wc][0]);
        const v16h b1 = lds_load_frag(&sB[cur][wc + 16][0]);
        acc00 = wmma_f16(a0, b0, acc00);
        acc01 = wmma_f16(a0, b1, acc01);
        acc10 = wmma_f16(a1, b0, acc10);
        acc11 = wmma_f16(a1, b1, acc11);
        cur ^= 1;
    }

    const int colL   = lane & 15;
    const int rowAdd = (lane & 16) ? 8 : 0;
#pragma unroll
    for (int i = 0; i < 2; ++i) {
#pragma unroll
        for (int j = 0; j < 2; ++j) {
            const v8f acc = (i == 0) ? (j == 0 ? acc00 : acc01)
                                     : (j == 0 ? acc10 : acc11);
#pragma unroll
            for (int r = 0; r < 8; ++r) {
                const int q = tileM + wr + i * 16 + rowAdd + r;
                const int d = wc + j * 16 + colL;               // 0..63
                Oheads[((size_t)b * LQ + q) * DMODEL + h * DH + d] = acc[r];
            }
        }
    }
}

// ---------------------------------------------------------------------------
extern "C" void kernel_launch(void* const* d_in, const int* in_sizes, int n_in,
                              void* d_out, int out_size, void* d_ws, size_t ws_size,
                              hipStream_t stream) {
    const float* query = (const float*)d_in[0];
    const float* key   = (const float*)d_in[1];
    const float* value = (const float*)d_in[2];
    const int*   mask  = (const int*)d_in[3];
    const float* bias  = (const float*)d_in[4];
    const float* Wq    = (const float*)d_in[5];
    const float* bq    = (const float*)d_in[6];
    const float* Wk    = (const float*)d_in[7];
    const float* bk    = (const float*)d_in[8];
    const float* Wv    = (const float*)d_in[9];
    const float* bv    = (const float*)d_in[10];
    const float* Wo    = (const float*)d_in[11];
    const float* bo    = (const float*)d_in[12];

    float* outO = (float*)d_out;
    float* attn = (float*)d_out + (size_t)B_SZ * LQ * DMODEL;

    _Float16* Qh = (_Float16*)d_ws;                                // 2*16*1024*64
    _Float16* Kh = Qh + (size_t)B_SZ * NHEADS * LQ * DH;           // 2*16*2048*64
    _Float16* Vt = Kh + (size_t)B_SZ * NHEADS * LK * DH;           // 2*16*64*2048
    float* Oheads = (float*)(Vt + (size_t)B_SZ * NHEADS * DH * LK);

    const dim3 blk(128);

    // Q projection -> f16 [B,H,Lq,dh]   (L = 1024, shift 10)
    gemm_proj<1, 10><<<dim3((B_SZ * LQ) / 64, DMODEL / 64, 1), blk, 0, stream>>>(
        query, Wq, bq, nullptr, Qh, B_SZ * LQ, DMODEL, DMODEL);
    // K projection -> f16 [B,H,Lk,dh]   (L = 2048, shift 11)
    gemm_proj<1, 11><<<dim3((B_SZ * LK) / 64, DMODEL / 64, 1), blk, 0, stream>>>(
        key, Wk, bk, nullptr, Kh, B_SZ * LK, DMODEL, DMODEL);
    // V projection -> f16 [B,H,dh,Lk]   (L = 2048, shift 11)
    gemm_proj<3, 11><<<dim3((B_SZ * LK) / 64, DMODEL / 64, 1), blk, 0, stream>>>(
        value, Wv, bv, nullptr, Vt, B_SZ * LK, DMODEL, DMODEL);

    // Scores + bias + mask -> attn region (pre-softmax, f32)
    scores_kernel<<<dim3(LQ / 64, LK / 64, B_SZ * NHEADS), blk, 0, stream>>>(
        Qh, Kh, bias, mask, attn);

    // Row softmax in place
    softmax_rows<<<dim3(B_SZ * NHEADS * LQ), dim3(256), 0, stream>>>(attn);

    // attn @ V -> heads-merged f32 [B,Lq,D]
    av_kernel<<<dim3(LQ / 64, 1, B_SZ * NHEADS), blk, 0, stream>>>(attn, Vt, Oheads);

    // Output projection -> d_out
    gemm_proj<0, 10><<<dim3((B_SZ * LQ) / 64, DMODEL / 64, 1), blk, 0, stream>>>(
        Oheads, Wo, bo, outO, nullptr, B_SZ * LQ, DMODEL, DMODEL);
}
